// CTCBeamDecoder_25701084299496
// MI455X (gfx1250) — compile-verified, matrix operand order checked by reading, not run on hardware
//
#include <hip/hip_runtime.h>
#include <hip/hip_bf16.h>
#include <stdint.h>

// CTC prefix beam search, one workgroup (256 threads, 8 wave32) per batch row.
// T=2000 sequential steps; beams/backpointers live in LDS; probs rows are
// double-buffered in LDS via TDM tensor_load_to_lds; e = p_tot (x) probs built
// with V_WMMA_F32_16X16X4_F32 (exact f32 outer product).

#define T_DIM 2000
#define B_DIM 64
#define S_DIM 128
#define W_BEAM 16
#define NCAND 2032            // W * (S-1)
#define HASH_MULT_ 1000003u
#define PRUNE_TH 0.001f
#define SMEM_BYTES 74368

typedef float     v8f  __attribute__((ext_vector_type(8)));
typedef float     v2f  __attribute__((ext_vector_type(2)));
typedef _Float16  v16h __attribute__((ext_vector_type(16)));
typedef unsigned  v4u  __attribute__((ext_vector_type(4)));
typedef int       v4i  __attribute__((ext_vector_type(4)));
typedef int       v8i  __attribute__((ext_vector_type(8)));

#if defined(__has_builtin)
#if __has_builtin(__builtin_amdgcn_tensor_load_to_lds) && __has_builtin(__builtin_amdgcn_s_wait_tensorcnt)
#define HAVE_TDM 1
#endif
#if __has_builtin(__builtin_amdgcn_wmma_f32_16x16x4_f32)
#define HAVE_WMMA_F32X4 1
#endif
#endif

__device__ __forceinline__ float pruneP(float v) { return v > PRUNE_TH ? v : 0.0f; }

// Sort key: larger value wins, then smaller index wins; key==0 is "empty".
__device__ __forceinline__ unsigned long long mkkey(float v, int idx) {
  unsigned u = __float_as_uint(v);
  u = (u & 0x80000000u) ? ~u : (u | 0x80000000u);
  return ((unsigned long long)u << 32) | (unsigned long long)(unsigned)(2047 - idx);
}

#ifdef HAVE_TDM
// 1D TDM copy: 128 contiguous f32 (512B) global -> LDS. D# per ISA 08 §8.
__device__ __forceinline__ void tdm_load_row(const float* gsrc, unsigned lds_byte_addr) {
  unsigned long long ga = (unsigned long long)(uintptr_t)gsrc;
  v4u g0;
  g0.x = 1u;                                                   // count=1, user
  g0.y = lds_byte_addr;                                        // lds_addr[63:32]
  g0.z = (unsigned)(ga & 0xFFFFFFFFull);                       // global_addr lo
  g0.w = (unsigned)((ga >> 32) & 0x01FFFFFFull) | (2u << 30);  // addr[56:32], type=2
  v8i g1;
  g1[0] = 0x20000;               // data_size=2 (4 bytes), mask=0, no flags
  g1[1] = (int)(S_DIM << 16);    // tensor_dim0 = 128   (bits 79:48)
  g1[2] = (int)(1 << 16);        // tensor_dim1 = 1     (bits 111:80)
  g1[3] = (int)(S_DIM << 16);    // tile_dim0   = 128   (bits 127:112)
  g1[4] = 1;                     // tile_dim1   = 1     (bits 143:128)
  g1[5] = S_DIM;                 // tensor_dim0_stride  (bits 207:160)
  g1[6] = 0;
  g1[7] = 0;
  v4i z4 = {0, 0, 0, 0};
#if __clang_major__ >= 23
  v8i z8 = {0, 0, 0, 0, 0, 0, 0, 0};
  __builtin_amdgcn_tensor_load_to_lds(g0, g1, z4, z4, z8, 0);
#else
  __builtin_amdgcn_tensor_load_to_lds(g0, g1, z4, z4, 0);
#endif
}
#endif

extern "C" __global__ __launch_bounds__(256) void ctc_beam_kernel(
    const float* __restrict__ x, const int* __restrict__ lengths,
    float* __restrict__ out) {
  extern __shared__ __align__(16) unsigned char smemRaw[];
  unsigned long long* slots = (unsigned long long*)(smemRaw);          // [3]
  float*    probsBuf = (float*)(smemRaw + 32);                         // [2][128]
  float*    eMat     = (float*)(smemRaw + 1056);                       // [16][128]
  unsigned* hS       = (unsigned*)(smemRaw + 9248);                    // [2][16]
  float*    pbS      = (float*)(smemRaw + 9376);                       // [2][16]
  float*    pnbS     = (float*)(smemRaw + 9504);                       // [2][16]
  int*      lastS    = (int*)(smemRaw + 9632);                         // [2][16]
  int*      lenS     = (int*)(smemRaw + 9760);                         // [2][16]
  float*    c0pbSh   = (float*)(smemRaw + 9888);                       // [16]
  float*    c0pnbSh  = (float*)(smemRaw + 9952);                       // [16]
  float*    maccSh   = (float*)(smemRaw + 10016);                      // [16]
  float*    upnbSh   = (float*)(smemRaw + 10080);                      // [16]
  float*    totsSh   = (float*)(smemRaw + 10144);                      // [16]
  int*      winIdx   = (int*)(smemRaw + 10208);                        // [16]
  unsigned short* bpL = (unsigned short*)(smemRaw + 10272);            // [2000][16]

  const int tid = threadIdx.x;
  const int b = blockIdx.x;
  int L = lengths[b];
  if (L < 1) L = 1;
  if (L > T_DIM) L = T_DIM;

  // Initial beam state: beam 0 = empty prefix (Pb=1); others sentinel hashes.
  if (tid < W_BEAM) {
    hS[tid]    = (tid == 0) ? 0u : (0x80000000u + (unsigned)tid);
    pbS[tid]   = (tid == 0) ? 1.0f : 0.0f;
    pnbS[tid]  = 0.0f;
    lenS[tid]  = 0;
    lastS[tid] = -1;
  }
#ifdef HAVE_TDM
  if (tid < 32) tdm_load_row(x + (size_t)b * S_DIM, (unsigned)(uintptr_t)probsBuf);
#endif
  __syncthreads();

  int cur = 0;
  for (int t = 0; t < L; ++t) {
    const int nxt = cur ^ 1;
    float* probsCur = probsBuf + (t & 1) * S_DIM;

    // ---------------- Phase 0 (wave 0 only) ----------------
    if (tid < 32) {
      const int lane = tid;
#ifdef HAVE_TDM
      {
        const int trow = (t + 1 < T_DIM) ? (t + 1) : (T_DIM - 1);
        tdm_load_row(x + ((size_t)trow * B_DIM + b) * S_DIM,
                     (unsigned)(uintptr_t)(probsBuf + ((t + 1) & 1) * S_DIM));
        __builtin_amdgcn_s_wait_tensorcnt(1);  // row t is complete
      }
#else
#pragma unroll
      for (int k = 0; k < 4; ++k)
        probsCur[lane + 32 * k] = x[((size_t)t * B_DIM + b) * S_DIM + lane + 32 * k];
#endif
      float ptv = 0.0f, pbv = 0.0f, pnbv = 0.0f;
      int lastv = -1;
      if (lane < W_BEAM) {
        pbv = pbS[cur * 16 + lane];
        pnbv = pnbS[cur * 16 + lane];
        ptv = pbv + pnbv;
        lastv = lastS[cur * 16 + lane];
        const float pblank = pruneP(probsCur[0]);
        c0pbSh[lane] = pblank * ptv;                                       // blank-extend
        c0pnbSh[lane] = (lastv >= 0) ? pruneP(probsCur[lastv]) * pnbv : 0.0f; // repeat-last
        maccSh[lane] = 0.0f;
      }
      if (lane == 0) { slots[0] = 0; slots[1] = 0; slots[2] = 0; }

      // e[b][c] = p_tot[b] * probs[c] via WMMA outer product (exact in f32:
      // one nonzero K term per output => identical to v_mul_f32).
      const float av = (lane < W_BEAM) ? ptv : 0.0f;
#ifdef HAVE_WMMA_F32X4
      v2f a; a.x = av; a.y = 0.0f;          // A[m,0]=p_tot[m]; K=1..3 zero
#pragma unroll
      for (int ch = 0; ch < 8; ++ch) {
        const float bval = (lane < 16) ? pruneP(probsCur[ch * 16 + lane]) : 0.0f;
        v2f bv; bv.x = bval; bv.y = 0.0f;   // B[0,n]=probs[n]; K=1..3 zero
        v8f c = {0.f, 0.f, 0.f, 0.f, 0.f, 0.f, 0.f, 0.f};
        v8f d = __builtin_amdgcn_wmma_f32_16x16x4_f32(false, a, false, bv,
                                                      (short)0, c, false, false);
        const int mbase = (lane >= 16) ? 8 : 0;
        const int n = ch * 16 + (lane & 15);
#pragma unroll
        for (int v = 0; v < 8; ++v) eMat[(mbase + v) * S_DIM + n] = d[v];
      }
#else
      v16h a16 = {}; a16[0] = (_Float16)av;
#pragma unroll
      for (int ch = 0; ch < 8; ++ch) {
        v16h b16 = {};
        b16[0] = (_Float16)((lane < 16) ? pruneP(probsCur[ch * 16 + lane]) : 0.0f);
        v8f c = {0.f, 0.f, 0.f, 0.f, 0.f, 0.f, 0.f, 0.f};
        v8f d = __builtin_amdgcn_wmma_f32_16x16x32_f16(false, a16, false, b16,
                                                       (short)0, c, false, false);
        const int mbase = (lane >= 16) ? 8 : 0;
        const int n = ch * 16 + (lane & 15);
#pragma unroll
        for (int v = 0; v < 8; ++v) eMat[(mbase + v) * S_DIM + n] = d[v];
      }
#endif
      // repeat-symbol correction: e[b][last[b]] uses p_b instead of p_tot
      if (lane < W_BEAM && lastv >= 1)
        eMat[lane * S_DIM + lastv] = pruneP(probsCur[lastv]) * pbv;
    }
    __syncthreads();  // B1: eMat / c0 / macc / slots ready for all waves

    // ---------------- Phase 1 (all 256 threads): 2032 extension candidates ----
    unsigned hreg[16];
#pragma unroll
    for (int w = 0; w < 16; ++w) hreg[w] = hS[cur * 16 + w];

    unsigned long long sk[9];
#pragma unroll
    for (int k = 0; k < 8; ++k) {
      const int j = tid + (k << 8);
      unsigned long long s = 0;
      if (j < NCAND) {
        const int bs = j / 127;
        const int c = j - bs * 127 + 1;                 // nonblank symbol 1..127
        const float e = eMat[bs * S_DIM + c];
        const unsigned eh = hreg[bs] * HASH_MULT_ + (unsigned)(c + 1);
        bool dup = false;
#pragma unroll
        for (int w = 0; w < 16; ++w)
          if (hreg[w] == eh) { dup = true; atomicAdd(&maccSh[w], e); }
        s = mkkey(dup ? -1.0f : e, 16 + j);
      }
      sk[k] = s;
    }
    sk[8] = 0;
    __syncthreads();  // B2: merge accumulation complete

    // ---------------- Phase 2 (wave 0): unchanged-prefix candidates ----------
    if (tid < W_BEAM) {
      const float up = c0pnbSh[tid] + maccSh[tid];
      upnbSh[tid] = up;
      sk[8] = mkkey(c0pbSh[tid] + up, tid);
    }

    // ---------------- Phase 3: stable top-16 (1 barrier/round, 3 slots) -----
    for (int r = 0; r < 16; ++r) {
      unsigned long long lm = 0;
#pragma unroll
      for (int k = 0; k < 9; ++k) lm = (sk[k] > lm) ? sk[k] : lm;
      if (lm) atomicMax(&slots[r % 3], lm);
      if (tid == 0) slots[(r + 1) % 3] = 0;  // prepare next round's slot
      __syncthreads();
      const unsigned long long win = slots[r % 3];
#pragma unroll
      for (int k = 0; k < 9; ++k)
        if (sk[k] == win) sk[k] = 0;         // consume (idx makes keys unique)
      if (tid == r) winIdx[r] = 2047 - (int)(unsigned)(win & 0xFFFFFFFFull);
    }
    __syncthreads();

    // ---------------- Phase 4 (wave 0): commit new beams + backpointers -----
    if (tid < W_BEAM) {
      const int w = tid;
      const int idx = winIdx[w];
      float npb, npnb; unsigned nh; int nlast, nlen, parent, sym;
      if (idx < W_BEAM) {
        npb = c0pbSh[idx]; npnb = upnbSh[idx];
        nh = hS[cur * 16 + idx]; nlast = lastS[cur * 16 + idx];
        nlen = lenS[cur * 16 + idx]; parent = idx; sym = -1;
      } else {
        const int j = idx - W_BEAM;
        const int bs = j / 127;
        const int c = j - bs * 127 + 1;
        const unsigned eh = hS[cur * 16 + bs] * HASH_MULT_ + (unsigned)(c + 1);
        bool dup = false;
#pragma unroll
        for (int w2 = 0; w2 < 16; ++w2) dup |= (hS[cur * 16 + w2] == eh);
        npb = 0.0f; npnb = dup ? 0.0f : eMat[bs * S_DIM + c];
        nh = eh; nlast = c; nlen = lenS[cur * 16 + bs] + 1;
        parent = bs; sym = c;
      }
      totsSh[w] = npb + npnb;
      hS[nxt * 16 + w] = nh;
      lastS[nxt * 16 + w] = nlast;
      lenS[nxt * 16 + w] = nlen;
      bpL[t * W_BEAM + w] =
          (unsigned short)(((unsigned)parent << 8) | ((unsigned)sym & 0xFFu));
      // rescale (wave-internal ordering: wave32 LDS ops are in-order)
      float mx = 1e-30f;
#pragma unroll
      for (int w2 = 0; w2 < 16; ++w2) mx = fmaxf(mx, totsSh[w2]);
      const float sc = 1.0f / mx;
      pbS[nxt * 16 + w] = npb * sc;
      pnbS[nxt * 16 + w] = npnb * sc;
    }
    cur = nxt;
  }

  // ---------------- Epilogue: outputs (prefs[B][T], lens[B], scores[B]) ------
  float* outPref = out + (size_t)b * T_DIM;
  for (int i = tid; i < T_DIM; i += 256) outPref[i] = -1.0f;
  __syncthreads();
  if (tid == 0) {
    const int len0 = lenS[cur * 16 + 0];
    out[(size_t)B_DIM * T_DIM + b] = (float)len0;
    out[(size_t)B_DIM * T_DIM + B_DIM + b] = pbS[cur * 16] + pnbS[cur * 16];
    int w = 0, pos = len0;
    for (int tt = L - 1; tt >= 0; --tt) {
      const unsigned v = bpL[tt * W_BEAM + w];
      const unsigned s8 = v & 0xFFu;
      if (s8 != 0xFFu) { --pos; if (pos >= 0 && pos < T_DIM) outPref[pos] = (float)s8; }
      w = (int)((v >> 8) & 15u);
    }
  }
}

extern "C" void kernel_launch(void* const* d_in, const int* in_sizes, int n_in,
                              void* d_out, int out_size, void* d_ws, size_t ws_size,
                              hipStream_t stream) {
  (void)in_sizes; (void)n_in; (void)out_size; (void)d_ws; (void)ws_size;
  const float* x = (const float*)d_in[0];
  const int* lengths = (const int*)d_in[1];
  float* out = (float*)d_out;
  (void)hipFuncSetAttribute((const void*)ctc_beam_kernel,
                            hipFuncAttributeMaxDynamicSharedMemorySize, SMEM_BYTES);
  ctc_beam_kernel<<<dim3(B_DIM), dim3(256), SMEM_BYTES, stream>>>(x, lengths, out);
}
